// GCN_61469571940701
// MI455X (gfx1250) — matrix operand order repaired and checked by reading, not compiled, not run on hardware
//
#include <hip/hip_runtime.h>
#include <hip/hip_bf16.h>

#define N_NODES   100000
#define N_EDGES   3200000
#define D         128
#define N_GRAPHS  256
#define N_CLASSES 2

// Padded LDS strides: 136 halves = 272 B = 68 dwords -> lane l hits bank (4*l)%64,
// all 16 lanes distinct (vs 64-dword stride = all lanes on one bank).
#define APITCH 136

// 6250 M-tiles total; each block owns 5 -> 1250 blocks, W staged once per block.
#define TILES_PER_BLOCK 5

typedef __attribute__((ext_vector_type(16))) _Float16 v16h;
typedef __attribute__((ext_vector_type(8)))  _Float16 v8h;
typedef __attribute__((ext_vector_type(8)))  float    v8f;

// ---------------- Kernel 1: degrees (deg_out[src]++, deg_in[dst]++) ----------------
__global__ void k_degrees(const int* __restrict__ ei,
                          float* __restrict__ deg_out,
                          float* __restrict__ deg_in) {
    int e = blockIdx.x * blockDim.x + threadIdx.x;
    if (e < N_EDGES) {
        int s = ei[e];
        int d = ei[N_EDGES + e];
        atomicAdd(&deg_out[s], 1.0f);
        atomicAdd(&deg_in[d], 1.0f);
    }
}

// ---------------- Kernel 2: in-place norm = clip(deg,1)^-1/2 ----------------
__global__ void k_norm(float* __restrict__ deg_out, float* __restrict__ deg_in) {
    int i = blockIdx.x * blockDim.x + threadIdx.x;
    if (i < N_NODES) {
        float a = deg_out[i]; a = a < 1.0f ? 1.0f : a;
        float b = deg_in[i];  b = b < 1.0f ? 1.0f : b;
        deg_out[i] = rsqrtf(a);   // becomes norm_src
        deg_in[i]  = rsqrtf(b);   // becomes norm_dst
    }
}

// ---------------- Kernel 3: SpMM scatter: agg[dst] += x[src] * norm_src[src] ----------
// One wave32 per edge; each lane handles 4 contiguous f32 (float4 gather, 4 atomics).
__global__ void __launch_bounds__(256) k_scatter(const float* __restrict__ x,
                                                 const int* __restrict__ ei,
                                                 const float* __restrict__ norm_src,
                                                 float* __restrict__ agg) {
    int wave = threadIdx.x >> 5;
    int lane = threadIdx.x & 31;
    int e = blockIdx.x * 8 + wave;
    if (e >= N_EDGES) return;
    int s = ei[e];
    int d = ei[N_EDGES + e];
    float ns = norm_src[s];
    const float4* xs = (const float4*)(x + (long long)s * D);
    float4 v = xs[lane];
    float* ad = agg + (long long)d * D + lane * 4;
    atomicAdd(ad + 0, v.x * ns);
    atomicAdd(ad + 1, v.y * ns);
    atomicAdd(ad + 2, v.z * ns);
    atomicAdd(ad + 3, v.w * ns);
}

// ---------------- Kernel 4: h = relu((agg * norm_dst) @ W + b), WMMA f16 ------------
// Block: 256 threads = 8 waves, persistent over TILES_PER_BLOCK 16-row M-tiles.
// W (f32->f16) is staged ONCE per block in LDS, pre-swizzled into B-fragment order
// sW[n][c][16] (c = k/16 half-wave chunk) so a B fragment is two contiguous 16B loads.
// Per tile: stage A (scaled by norm_dst), prefetch next tile, 4x WMMA, bias+relu store.
// h aliases agg: each block fully stages a tile's rows into LDS before storing them.
__global__ void __launch_bounds__(256) k_gemm_relu(const float* __restrict__ agg,
                                                   const float* __restrict__ norm_dst,
                                                   const float* __restrict__ W,
                                                   const float* __restrict__ bias,
                                                   float* __restrict__ h) {
    __shared__ __align__(32) _Float16 sW[D * APITCH];   // ~34 KB, [n][c][16] swizzled
    __shared__ __align__(32) _Float16 sA[16 * APITCH];  // ~4.3 KB, [m][k], padded pitch

    const int tid = threadIdx.x;

    // Stage W (f32 -> f16) with fragment swizzle. Global read coalesced over [k][n].
    #pragma unroll 4
    for (int i = 0; i < (D * D) / 256; ++i) {
        int idx = i * 256 + tid;
        int k   = idx >> 7;
        int n   = idx & (D - 1);
        sW[n * APITCH + (k >> 4) * 16 + (k & 15)] = (_Float16)W[idx];
    }

    const int wv   = tid >> 5;
    const int lane = tid & 31;
    const int l    = lane & 15;     // 0..15
    const int hi   = lane >> 4;     // half-wave select
    const int n0   = wv * 16;

    const _Float16* aRow = sA + l * APITCH;              // row M = l
    const _Float16* bCol = sW + (n0 + l) * APITCH;       // column N = n0 + l
    const int col  = n0 + l;
    const float bv = bias[col];

    for (int t = 0; t < TILES_PER_BLOCK; ++t) {
        const int m0 = (blockIdx.x * TILES_PER_BLOCK + t) * 16;

        // Stage A tile, scaled by norm_dst (8 iterations x 256 threads)
        #pragma unroll
        for (int i = 0; i < (16 * D) / 256; ++i) {
            int idx = i * 256 + tid;
            int r   = idx >> 7;         // row within tile 0..15
            int k   = idx & (D - 1);
            int row = m0 + r;
            sA[r * APITCH + k] = (_Float16)(agg[(long long)row * D + k] * norm_dst[row]);
        }
        __syncthreads();   // covers W staging (t==0) and A staging (every tile)

        // Prefetch next tile (8 KB): 256 threads x 32B stride covers every line.
        if (t + 1 < TILES_PER_BLOCK) {
            const char* nxt = (const char*)(agg + (long long)(m0 + 16) * D);
            __builtin_prefetch(nxt + tid * 32, 0, 3);
        }

        v8f c = {};
        #pragma unroll
        for (int kb = 0; kb < 4; ++kb) {
            const int k0 = kb * 32;
            // A fragment (contiguous K order, matching B): lane half 'hi' holds
            // K = {k0+16hi .. +15}: two contiguous 16B chunks.
            v8h alo = *(const v8h*)(aRow + k0 + hi * 16);
            v8h ahi = *(const v8h*)(aRow + k0 + hi * 16 + 8);
            v16h a = __builtin_shufflevector(alo, ahi,
                                             0, 1, 2, 3, 4, 5, 6, 7,
                                             8, 9, 10, 11, 12, 13, 14, 15);
            // B fragment: half-wave hi covers K-chunk c = 2*kb + hi, column l:
            // 16 contiguous halves in swizzled sW -> two 16B loads.
            const _Float16* bp = bCol + (kb * 2 + hi) * 16;
            v8h blo = *(const v8h*)(bp);
            v8h bhi = *(const v8h*)(bp + 8);
            v16h b = __builtin_shufflevector(blo, bhi,
                                             0, 1, 2, 3, 4, 5, 6, 7,
                                             8, 9, 10, 11, 12, 13, 14, 15);
            c = __builtin_amdgcn_wmma_f32_16x16x32_f16(false, a, false, b,
                                                       (short)0, c, false, false);
        }

        // C/D layout: VGPR r -> M = 8*hi + r, N = l. Add bias, relu, store.
        // Single base pointer; row stride 512B folds into the 24-bit inst offset,
        // so this lowers to one address pair + 8 immediate-offset stores.
        float* hp = h + ((long long)(m0 + hi * 8) * D + col);
        #pragma unroll
        for (int r = 0; r < 8; ++r) {
            float val = c[r] + bv;
            hp[r * D] = val > 0.0f ? val : 0.0f;
        }
        __syncthreads();   // sA reads done before next tile overwrites it
    }
}

// ---------------- Kernel 5: per-graph sum + counts ----------------
__global__ void k_pool(const float* __restrict__ h, const int* __restrict__ gid,
                       float* __restrict__ sums, float* __restrict__ counts) {
    int node = blockIdx.x;
    int d    = threadIdx.x;
    int g    = gid[node];
    atomicAdd(&sums[g * D + d], h[(long long)node * D + d]);
    if (d == 0) atomicAdd(&counts[g], 1.0f);
}

// ---------------- Kernel 6: mean + classifier (256 graphs x 2 classes) --------------
__global__ void k_classify(const float* __restrict__ sums, const float* __restrict__ counts,
                           const float* __restrict__ Wc, const float* __restrict__ bc,
                           float* __restrict__ out) {
    int g = threadIdx.x;            // 256 threads, one graph each
    float cnt = counts[g];
    float inv = 1.0f / (cnt < 1.0f ? 1.0f : cnt);
    float a0 = bc[0], a1 = bc[1];
    #pragma unroll 4
    for (int d = 0; d < D; ++d) {
        float p = sums[g * D + d] * inv;
        a0 += p * Wc[d * 2 + 0];
        a1 += p * Wc[d * 2 + 1];
    }
    out[g * 2 + 0] = a0;
    out[g * 2 + 1] = a1;
}

extern "C" void kernel_launch(void* const* d_in, const int* in_sizes, int n_in,
                              void* d_out, int out_size, void* d_ws, size_t ws_size,
                              hipStream_t stream) {
    const float* x   = (const float*)d_in[0];
    const int*   ei  = (const int*)d_in[1];
    const int*   gid = (const int*)d_in[2];
    const float* W   = (const float*)d_in[3];
    const float* b   = (const float*)d_in[4];
    const float* Wc  = (const float*)d_in[5];
    const float* bc  = (const float*)d_in[6];
    float* out = (float*)d_out;

    // Workspace layout (floats): agg[N*D] | deg_out[N] | deg_in[N] | sums[G*D] | counts[G]
    float* agg     = (float*)d_ws;
    float* deg_out = agg + (size_t)N_NODES * D;
    float* deg_in  = deg_out + N_NODES;
    float* sums    = deg_in + N_NODES;
    float* counts  = sums + N_GRAPHS * D;

    size_t zero_floats = (size_t)N_NODES * D + 2 * (size_t)N_NODES
                       + (size_t)N_GRAPHS * D + N_GRAPHS;
    hipMemsetAsync(d_ws, 0, zero_floats * sizeof(float), stream);

    k_degrees<<<(N_EDGES + 255) / 256, 256, 0, stream>>>(ei, deg_out, deg_in);
    k_norm<<<(N_NODES + 255) / 256, 256, 0, stream>>>(deg_out, deg_in);
    k_scatter<<<N_EDGES / 8, 256, 0, stream>>>(x, ei, deg_out, agg);
    k_gemm_relu<<<(N_NODES / 16) / TILES_PER_BLOCK, 256, 0, stream>>>(agg, deg_in, W, b, agg);
    k_pool<<<N_NODES, D, 0, stream>>>(agg, gid, sums, counts);
    k_classify<<<1, N_GRAPHS, 0, stream>>>(sums, counts, Wc, bc, out);
}
